// CAAN_13408887898592
// MI455X (gfx1250) — matrix operand order repaired and checked by reading, not compiled, not run on hardware
//
#include <hip/hip_runtime.h>

// ---------------------------------------------------------------------------
// CAAN attention for MI455X (gfx1250, wave32, WMMA).
//   d_in: R, Wq, bq, Wk, bk, Wv, bv, W1, b1, W2, b2   (all f32)
//   d_out: winner_score [8, 2048] f32
//   d_ws : needs ~50 MB (f16 Wq/Wk/Wv + f16 Q, K, V^T)
// ---------------------------------------------------------------------------

typedef __attribute__((ext_vector_type(16))) _Float16 v16h;
typedef __attribute__((ext_vector_type(8)))  _Float16 v8h;
typedef __attribute__((ext_vector_type(8)))  float    v8f;

#define WMMA_F16(a, b, c) \
  __builtin_amdgcn_wmma_f32_16x16x32_f16(false, (a), false, (b), (short)0, (c), false, false)

constexpr int   NQ    = 2048;   // sequence length
constexpr int   D     = 512;    // query/caan dim
constexpr int   BATCH = 8;
constexpr int   NHID  = 30;

// workspace layout, in _Float16 elements
constexpr size_t WH_OFF = 0;                          // 3 x [512][512] f16 weights
constexpr size_t QH_OFF = WH_OFF + (size_t)3 * D * D; // Q  f16 [8][2048][512]
constexpr size_t MAT_SZ = (size_t)BATCH * NQ * D;
constexpr size_t KH_OFF = QH_OFF + MAT_SZ;            // K  f16 [8][2048][512]
constexpr size_t VT_OFF = KH_OFF + MAT_SZ;            // V^T f16 [8][512][2048]

// ---------------------------------------------------------------------------
// Fragment loaders for wave32 WMMA f16 layouts (CDNA5 ISA 7.12.2).
// A (16xK tile, row-major, stride ld): lane l -> row m=l&15;
//   halves 0..7 = K kb+0..7, 8..15 = K kb+16..23, kb = 8*(l>>4).
// B (Kx16 tile given as the TRANSPOSED row-major matrix W[n][k], stride ld):
//   lane l -> col n=l&15; halves 0..7 = K kb+0..7, 8..15 = K kb+16..23, kb = 8*(l>>4).
// C/D f32: lane l -> col n=l&15; c[j] -> row m = j + 8*(l>>4).
// ---------------------------------------------------------------------------
__device__ __forceinline__ v16h load_a_frag(const _Float16* A, int ld) {
  const int l = threadIdx.x & 31;
  const _Float16* p = A + (size_t)(l & 15) * ld + ((l >> 4) << 3);
  v8h lo = *(const v8h*)p;          // K kb+0..7   (16B aligned)
  v8h hi = *(const v8h*)(p + 16);   // K kb+16..23
  return __builtin_shufflevector(lo, hi, 0, 1, 2, 3, 4, 5, 6, 7,
                                 8, 9, 10, 11, 12, 13, 14, 15);
}

__device__ __forceinline__ v16h load_bT_frag(const _Float16* W, int ld) {
  const int l = threadIdx.x & 31;
  const _Float16* p = W + (size_t)(l & 15) * ld + ((l >> 4) << 3);
  v8h lo = *(const v8h*)p;
  v8h hi = *(const v8h*)(p + 16);
  return __builtin_shufflevector(lo, hi, 0, 1, 2, 3, 4, 5, 6, 7,
                                 8, 9, 10, 11, 12, 13, 14, 15);
}

// ---------------------------------------------------------------------------
// Kernel 1: convert Wq/Wk/Wv f32 -> f16 (L2-resident, reused 2048x each).
// ---------------------------------------------------------------------------
__global__ void __launch_bounds__(256) wcvt_kernel(const float* __restrict__ Wq,
                                                   const float* __restrict__ Wk,
                                                   const float* __restrict__ Wv,
                                                   _Float16* __restrict__ Wh) {
  const int idx = blockIdx.x * 256 + threadIdx.x;     // < 3*D*D
  const int mat = idx / (D * D);
  const int off = idx - mat * (D * D);
  const float* src = (mat == 0) ? Wq : (mat == 1) ? Wk : Wv;
  Wh[idx] = (_Float16)src[off];
}

// ---------------------------------------------------------------------------
// Kernel 2: Q/K/V projections.  grid = (128 row tiles, 4 dout groups, 8*3).
// Block = 256 threads (8 waves).  R tile (16x512) staged in LDS as f16,
// each wave owns one 16-wide dout tile: 16 x v_wmma_f32_16x16x32_f16.
// V is stored transposed so the attention PV B-operand is contiguous.
// ---------------------------------------------------------------------------
__global__ void __launch_bounds__(256) proj_kernel(const float* __restrict__ R,
                                                   const float* __restrict__ bq,
                                                   const float* __restrict__ bk,
                                                   const float* __restrict__ bv,
                                                   _Float16* __restrict__ ws) {
  const int n0  = blockIdx.x * 16;
  const int d0  = blockIdx.y * 128;
  const int mat = blockIdx.z % 3;
  const int bat = blockIdx.z / 3;
  const int w   = threadIdx.x >> 5;
  const int l   = threadIdx.x & 31;
  const int n   = l & 15;
  const int hi  = l >> 4;

  __shared__ _Float16 sA[16 * D];                     // 16 KB

  const float* Rt = R + ((size_t)bat * NQ + n0) * D;  // tile is contiguous
  for (int i = threadIdx.x; i < 16 * D; i += 256) sA[i] = (_Float16)Rt[i];
  __syncthreads();

  const _Float16* Wmat = ws + WH_OFF + (size_t)mat * D * D;
  const int dcol = d0 + w * 16;

  v8f c = {};
#pragma unroll
  for (int kk = 0; kk < D / 32; ++kk) {
    v16h a = load_a_frag(sA + kk * 32, D);
    v16h b = load_bT_frag(Wmat + (size_t)dcol * D + kk * 32, D);
    c = WMMA_F16(a, b, c);
  }

  const float* bias = (mat == 0) ? bq : (mat == 1) ? bk : bv;
  const float bb = bias[dcol + n];

  if (mat < 2) {                                      // Q,K row-major [n][d]
    _Float16* out = ws + (mat == 0 ? QH_OFF : KH_OFF) +
                    ((size_t)bat * NQ + n0) * D;
#pragma unroll
    for (int j = 0; j < 8; ++j)
      out[(size_t)(j + 8 * hi) * D + dcol + n] = (_Float16)(c[j] + bb);
  } else {                                            // V transposed [d][n]
    _Float16* out = ws + VT_OFF + (size_t)bat * D * NQ;
#pragma unroll
    for (int j = 0; j < 8; ++j)
      out[(size_t)(dcol + n) * NQ + n0 + j + 8 * hi] = (_Float16)(c[j] + bb);
  }
}

// ---------------------------------------------------------------------------
// Kernel 3: fused attention + head.  grid = (32 query tiles, 8 batches).
// Block = 256 threads (8 waves) owns a 64-query tile; ~148 KB LDS
// (sQ 64K + sK 64K + sP 8K + sums/head) -- sized for the 320 KB CDNA5 WGP.
// Per 64-key block:
//   S phase : waves form a 4x2 grid over the 4x4 S-tile grid; each wave does
//             2 chained WMMAs per k-step (A reused, K-frags shared via LDS).
//   PV phase: each wave owns a 64-wide d-slice with a 4x4 v8f accumulator
//             grid: 16 WMMAs per 8 fragment loads.
// Epilogue: shuffle+LDS-atomic row sums, normalize, f16 'a' tile overlays the
// dead sQ region, then the 512->30->1 head is computed in-block.
// ---------------------------------------------------------------------------
__global__ void __launch_bounds__(256) attn_kernel(const _Float16* __restrict__ ws,
                                                   const float* __restrict__ W1,
                                                   const float* __restrict__ b1,
                                                   const float* __restrict__ W2,
                                                   const float* __restrict__ b2,
                                                   float* __restrict__ out) {
  const int n0  = blockIdx.x * 64;
  const int bat = blockIdx.y;
  const int tid = threadIdx.x;
  const int w   = tid >> 5;
  const int l   = tid & 31;
  const int n   = l & 15;
  const int hi  = l >> 4;
  const int mi  = w >> 1;        // 0..3 : S-tile row group for this wave
  const int ng  = (w & 1) * 2;   // 0/2  : first of two S-tile columns

  __shared__ __align__(16) unsigned char smem[147712];
  _Float16* sQ   = (_Float16*)smem;                    // [64][512]  64 KB
  _Float16* sK   = (_Float16*)(smem + 65536);          // [64][512]  64 KB
  _Float16* sP   = (_Float16*)(smem + 131072);         // [64][64]    8 KB
  float*    sSum = (float*)   (smem + 139264);         // [64]
  float*    sH   = (float*)   (smem + 139520);         // [64][32]    8 KB
  _Float16* aH   = sQ;           // 'a' tile reuses sQ after the key loop

  // ---- stage Q tile (64x512 f16, contiguous) ----
  {
    const uint4* src = (const uint4*)(ws + QH_OFF + ((size_t)bat * NQ + n0) * D);
    uint4* dst = (uint4*)sQ;
    for (int i = tid; i < 64 * D / 8; i += 256) dst[i] = src[i];
  }
  if (tid < 64) sSum[tid] = 0.f;

  const _Float16* Kb = ws + KH_OFF + (size_t)bat * NQ * D;
  const _Float16* Vt = ws + VT_OFF + (size_t)bat * D * NQ;

  v8f acc[4][4];
  {
    v8f z = {};
#pragma unroll
    for (int r = 0; r < 4; ++r)
#pragma unroll
      for (int t = 0; t < 4; ++t) acc[r][t] = z;
  }
  float rsum[8] = {0.f, 0.f, 0.f, 0.f, 0.f, 0.f, 0.f, 0.f};
  const float scale = 0.044194173824159216f;           // 1/sqrt(512)

  for (int kb = 0; kb < NQ; kb += 64) {
    __syncthreads();               // previous iteration's sK/sP reads done
    // ---- stage K block (64x512 f16) ----
    {
      const uint4* src = (const uint4*)(Kb + (size_t)kb * D);
      uint4* dst = (uint4*)sK;
      for (int i = tid; i < 64 * D / 8; i += 256) dst[i] = src[i];
    }
    if (kb + 64 < NQ)              // pull next K block toward L2/L0
      __builtin_prefetch(Kb + (size_t)(kb + 64) * D + (size_t)tid * 128, 0, 3);
    __syncthreads();               // sK ready

    // ---- S phase: two 16x16 tiles, A fragment reused ----
    v8f s0 = {}, s1 = {};
#pragma unroll
    for (int kk = 0; kk < D / 32; ++kk) {
      v16h a  = load_a_frag(sQ + (size_t)mi * 16 * D + kk * 32, D);
      v16h b0 = load_bT_frag(sK + (size_t)(ng + 0) * 16 * D + kk * 32, D);
      v16h b1 = load_bT_frag(sK + (size_t)(ng + 1) * 16 * D + kk * 32, D);
      s0 = WMMA_F16(a, b0, s0);
      s1 = WMMA_F16(a, b1, s1);
    }
    // ---- exp + running row sums (reference: plain exp, no max-subtract) ----
#pragma unroll
    for (int j = 0; j < 8; ++j) {
      const float e0 = __expf(s0[j] * scale);
      const float e1 = __expf(s1[j] * scale);
      rsum[j] += e0 + e1;
      const int m = mi * 16 + j + 8 * hi;
      sP[m * 64 + (ng + 0) * 16 + n] = (_Float16)e0;
      sP[m * 64 + (ng + 1) * 16 + n] = (_Float16)e1;
    }
    __syncthreads();               // sP tile visible to all waves

    // ---- PV phase: 4x4 accumulator grid over this wave's 64-wide d slice ----
#pragma unroll
    for (int kk = 0; kk < 2; ++kk) {
      v16h af[4];
#pragma unroll
      for (int r = 0; r < 4; ++r)
        af[r] = load_a_frag(sP + (size_t)r * 16 * 64 + kk * 32, 64);
#pragma unroll
      for (int t = 0; t < 4; ++t) {
        v16h b = load_bT_frag(Vt + (size_t)(w * 64 + t * 16) * NQ + kb + kk * 32, NQ);
#pragma unroll
        for (int r = 0; r < 4; ++r)
          acc[r][t] = WMMA_F16(af[r], b, acc[r][t]);
      }
    }
  }

  // ---- row sums: reduce 16 lanes per half-wave, combine waves via ds_add ----
#pragma unroll
  for (int j = 0; j < 8; ++j) {
    float v = rsum[j];
    v += __shfl_xor(v, 1, 32);
    v += __shfl_xor(v, 2, 32);
    v += __shfl_xor(v, 4, 32);
    v += __shfl_xor(v, 8, 32);
    if (n == 0) atomicAdd(&sSum[mi * 16 + j + 8 * hi], v);
  }
  __syncthreads();

  // ---- normalize; stage 'a' tile (f16) into the dead sQ region ----
#pragma unroll
  for (int r = 0; r < 4; ++r)
#pragma unroll
    for (int j = 0; j < 8; ++j) {
      const int m = r * 16 + j + 8 * hi;
      const float inv = 1.0f / sSum[m];
#pragma unroll
      for (int t = 0; t < 4; ++t)
        aH[(size_t)m * D + w * 64 + t * 16 + n] = (_Float16)(acc[r][t][j] * inv);
    }
  __syncthreads();

  // ---- head: h = a.W1^T + b1  (64x30 dots of length 512) ----
  for (int idx = tid; idx < 64 * NHID; idx += 256) {
    const int m = idx / NHID;
    const int h = idx - m * NHID;
    const float* w1 = W1 + (size_t)h * D;
    const _Float16* ar = aH + (size_t)m * D;
    float a2 = b1[h];
#pragma unroll 8
    for (int d = 0; d < D; ++d) a2 += (float)ar[d] * w1[d];
    sH[m * 32 + h] = a2;
  }
  __syncthreads();

  // ---- score = h.W2 + b2 ----
  if (tid < 64) {
    float sc = b2[0];
#pragma unroll
    for (int h = 0; h < NHID; ++h) sc += sH[tid * 32 + h] * W2[h];
    out[(size_t)bat * NQ + n0 + tid] = sc;
  }
}

// ---------------------------------------------------------------------------
extern "C" void kernel_launch(void* const* d_in, const int* in_sizes, int n_in,
                              void* d_out, int out_size, void* d_ws, size_t ws_size,
                              hipStream_t stream) {
  const float* R  = (const float*)d_in[0];
  const float* Wq = (const float*)d_in[1];
  const float* bq = (const float*)d_in[2];
  const float* Wk = (const float*)d_in[3];
  const float* bk = (const float*)d_in[4];
  const float* Wv = (const float*)d_in[5];
  const float* bv = (const float*)d_in[6];
  const float* W1 = (const float*)d_in[7];
  const float* b1 = (const float*)d_in[8];
  const float* W2 = (const float*)d_in[9];
  const float* b2 = (const float*)d_in[10];

  _Float16* ws = (_Float16*)d_ws;   // needs (3*512*512 + 3*8*2048*512)*2 B ~= 50 MB

  wcvt_kernel<<<(3 * D * D) / 256, 256, 0, stream>>>(Wq, Wk, Wv, ws + WH_OFF);
  proj_kernel<<<dim3(NQ / 16, D / 128, BATCH * 3), 256, 0, stream>>>(R, bq, bk, bv, ws);
  attn_kernel<<<dim3(NQ / 64, BATCH), 256, 0, stream>>>(ws, W1, b1, W2, b2,
                                                        (float*)d_out);
}
